// NodeModel_14542759264797
// MI455X (gfx1250) — matrix operand order repaired and checked by reading, not compiled, not run on hardware
//
#include <hip/hip_runtime.h>

#define NN 20000
#define EE 80000
#define DD 1024
#define LDH 1032  // LDS h1 row stride in bf16 elements (1024 + 8 skew)
#define ASTR 136  // staged A-tile row stride in bf16 elements (128 + 8 skew)
#define CK 128    // K-chunk staged into LDS per barrier interval

typedef __bf16 bf16_t;
typedef __attribute__((ext_vector_type(16))) __bf16 v16bf;
typedef __attribute__((ext_vector_type(8)))  __bf16 v8bf;
typedef __attribute__((ext_vector_type(4)))  __bf16 v4bf;
typedef __attribute__((ext_vector_type(8)))  float  v8f;
typedef __attribute__((ext_vector_type(4)))  float  f4;
typedef int i32x4 __attribute__((vector_size(16)));

#if __has_builtin(__builtin_amdgcn_global_load_async_to_lds_b128)
#define ASYNC_LDS 1
#else
#define ASYNC_LDS 0
#endif

// ---------------------------------------------------------------------------
// Async global->LDS 16-byte copy (ASYNCcnt-tracked DMA path on gfx1250),
// with a synchronous reg-bounce fallback of identical structure.
// Probe-learned signature: both pointers are generic `int4*`.
// ---------------------------------------------------------------------------
__device__ __forceinline__ void copy16_g2l(const bf16_t* g, bf16_t* l) {
#if ASYNC_LDS
  __builtin_amdgcn_global_load_async_to_lds_b128(
      (i32x4*)g, (i32x4*)l, 0, 0);
#else
  *reinterpret_cast<v8bf*>(l) = *reinterpret_cast<const v8bf*>(g);
#endif
}

__device__ __forceinline__ void async_wait0() {
#if ASYNC_LDS
#if __has_builtin(__builtin_amdgcn_s_wait_asynccnt)
  __builtin_amdgcn_s_wait_asynccnt(0);
#else
  asm volatile("s_wait_asynccnt 0x0" ::: "memory");
#endif
#endif
}

// ---------------------------------------------------------------------------
// WMMA wrapper: D = A(16x32 bf16) * B(32x16 bf16) + C(16x16 f32)
// ---------------------------------------------------------------------------
__device__ __forceinline__ v8f wmma_bf16(v16bf a, v16bf b, v8f c) {
  return __builtin_amdgcn_wmma_f32_16x16x32_bf16(
      /*neg_a=*/false, a, /*neg_b=*/false, b,
      /*c_mod=*/(short)0, c, /*reuse_a=*/false, /*reuse_b=*/false);
}

__device__ __forceinline__ v16bf make_v16(v8bf lo, v8bf hi) {
  v16bf r;
#pragma unroll
  for (int i = 0; i < 8; ++i) { r[i] = lo[i]; r[i + 8] = hi[i]; }
  return r;
}

// A fragment (16x32 bf16, row-major K contiguous).
// ISA layout: lanes 0-15 -> K {0..7, 16..23}; lanes 16-31 -> K {8..15, 24..31}
__device__ __forceinline__ v16bf load_a_bf16(const bf16_t* p, int lane) {
  const int koff = (lane & 16) ? 8 : 0;
  v8bf lo = *reinterpret_cast<const v8bf*>(p + koff);
  v8bf hi = *reinterpret_cast<const v8bf*>(p + koff + 16);
  return make_v16(lo, hi);
}

// B fragment (32x16 bf16). Weights stored transposed: Wt[n][k] (K contiguous).
// ISA layout: lanes 0-15 -> K {0..7, 16..23} (N=lane); lanes 16-31 -> K {8..15, 24..31}.
__device__ __forceinline__ v16bf load_b_bf16(const bf16_t* p, int lane) {
  const int koff = (lane & 16) ? 8 : 0;
  v8bf lo = *reinterpret_cast<const v8bf*>(p + koff);
  v8bf hi = *reinterpret_cast<const v8bf*>(p + koff + 16);
  return make_v16(lo, hi);
}

// ---------------------------------------------------------------------------
// Prep kernels
// ---------------------------------------------------------------------------
__global__ void transpose_to_bf16(const float* __restrict__ w,
                                  bf16_t* __restrict__ wt, int K, int Nc) {
  const int i = blockIdx.x * blockDim.x + threadIdx.x;
  if (i >= K * Nc) return;
  const int k = i / Nc;
  const int n = i - k * Nc;
  wt[(size_t)n * K + k] = (__bf16)w[i];
}

__global__ void f32_to_bf16_v4(const float* __restrict__ in,
                               bf16_t* __restrict__ out, int n4) {
  const int i = blockIdx.x * blockDim.x + threadIdx.x;
  if (i >= n4) return;
  f4 v = reinterpret_cast<const f4*>(in)[i];
  v4bf o;
  o[0] = (__bf16)v.x; o[1] = (__bf16)v.y; o[2] = (__bf16)v.z; o[3] = (__bf16)v.w;
  reinterpret_cast<v4bf*>(out)[i] = o;
}

__global__ void zero_f32(float* __restrict__ p, int n) {
  const int i = blockIdx.x * blockDim.x + threadIdx.x;
  if (i < n) p[i] = 0.0f;
}

// mean = sums / max(counts,1), emitted directly as bf16 (sums left untouched)
__global__ void mean_to_bf16(const float* __restrict__ sums,
                             const float* __restrict__ counts,
                             bf16_t* __restrict__ meanb, int total) {
  const int i = blockIdx.x * blockDim.x + threadIdx.x;
  if (i < total) meanb[i] = (__bf16)(sums[i] / fmaxf(counts[i >> 10], 1.0f));
}

// ---------------------------------------------------------------------------
// Edge MLP: h = relu([x[row] | eattr] @ W1a + b1a) @ W1b + b1b, scatter-add.
// 8 waves/block, 64 edges/block; wave tile = 4 M x 2 N (8 WMMA accumulators).
// Layer-1 A tiles are cooperatively async-staged into LDS (8x vmem cut);
// layer-1 output (64x1024 bf16) also lives in LDS. 149.5 KB LDS -> two
// blocks co-resident per 320 KB WGP for cross-block pipelining.
// ---------------------------------------------------------------------------
__global__ void __launch_bounds__(256) edge_mlp_kernel(
    const bf16_t* __restrict__ xb, const long long* __restrict__ ei,
    const bf16_t* __restrict__ eb,
    const bf16_t* __restrict__ w1a_t, const float* __restrict__ b1a,
    const bf16_t* __restrict__ w1b_t, const float* __restrict__ b1b,
    float* __restrict__ sums, float* __restrict__ counts) {
  __shared__ int s_row[64];
  __shared__ int s_col[64];
  extern __shared__ bf16_t lds[];
  bf16_t* h1   = lds;                 // [64][LDH]
  bf16_t* abuf = lds + 64 * LDH;      // [64][ASTR] staged A chunk

  const int tid   = threadIdx.x;
  const int lane  = tid & 31;
  const int wv    = tid >> 5;
  const int nlane = lane & 15;
  const int mhi   = (lane & 16) ? 8 : 0;
  const int e0    = blockIdx.x * 64;

  if (tid < 64) {
    const int e = e0 + tid;
    s_row[tid] = (int)ei[e];
    const int c = (int)ei[EE + e];
    s_col[tid] = c;
    atomicAdd(&counts[c], 1.0f);
  }
  __syncthreads();

  // ---------------- layer 1: [64,2048] x [2048,1024] -> LDS ----------------
  for (int np = 0; np < 4; ++np) {
    const int nb0 = np * 256 + wv * 32;
    const int nb1 = nb0 + 16;
    v8f acc0[4], acc1[4];
    {
      const float bv0 = b1a[nb0 + nlane];
      const float bv1 = b1a[nb1 + nlane];
#pragma unroll
      for (int mt = 0; mt < 4; ++mt)
#pragma unroll
        for (int r = 0; r < 8; ++r) { acc0[mt][r] = bv0; acc1[mt][r] = bv1; }
    }
    const bf16_t* wrow0 = w1a_t + (size_t)(nb0 + nlane) * 2048;
    const bf16_t* wrow1 = w1a_t + (size_t)(nb1 + nlane) * 2048;

    for (int kc = 0; kc < 2048; kc += CK) {
      // Cooperative async stage of A[64][CK] (gathered x rows, then eattr)
#pragma unroll
      for (int t = 0; t < 4; ++t) {
        const int i   = tid + t * 256;  // 0..1023 = 64 rows x 16 chunks of 16B
        const int row = i >> 4;
        const int c   = (i & 15) * 8;
        const bf16_t* src = (kc < 1024)
            ? xb + (size_t)s_row[row] * DD + kc + c
            : eb + (size_t)(e0 + row) * DD + (kc - 1024) + c;
        copy16_g2l(src, abuf + row * ASTR + c);
      }
      async_wait0();
      __syncthreads();

#pragma unroll
      for (int k2 = 0; k2 < CK; k2 += 32) {
        const int k = kc + k2;
        __builtin_prefetch(wrow0 + k + 512, 0, 1);
        __builtin_prefetch(wrow1 + k + 512, 0, 1);
        const v16bf b0 = load_b_bf16(wrow0 + k, lane);
        const v16bf b1 = load_b_bf16(wrow1 + k, lane);
        v16bf a[4];
#pragma unroll
        for (int mt = 0; mt < 4; ++mt)
          a[mt] = load_a_bf16(abuf + (mt * 16 + nlane) * ASTR + k2, lane);
#pragma unroll
        for (int mt = 0; mt < 4; ++mt) acc0[mt] = wmma_bf16(a[mt], b0, acc0[mt]);
#pragma unroll
        for (int mt = 0; mt < 4; ++mt) acc1[mt] = wmma_bf16(a[mt], b1, acc1[mt]);
      }
      __syncthreads();  // all waves done reading abuf before next stage
    }
#pragma unroll
    for (int mt = 0; mt < 4; ++mt)
#pragma unroll
      for (int r = 0; r < 8; ++r) {
        const int mrow = mt * 16 + r + mhi;
        h1[(size_t)mrow * LDH + nb0 + nlane] = (__bf16)fmaxf(acc0[mt][r], 0.0f);
        h1[(size_t)mrow * LDH + nb1 + nlane] = (__bf16)fmaxf(acc1[mt][r], 0.0f);
      }
  }
  __syncthreads();

  // ---------------- layer 2: [64,1024] x [1024,1024] -> atomics ------------
  const bf16_t* arow_h[4];
#pragma unroll
  for (int mt = 0; mt < 4; ++mt)
    arow_h[mt] = h1 + (size_t)(mt * 16 + nlane) * LDH;

  for (int np = 0; np < 4; ++np) {
    const int nb0 = np * 256 + wv * 32;
    const int nb1 = nb0 + 16;
    v8f acc0[4], acc1[4];
    {
      const float bv0 = b1b[nb0 + nlane];
      const float bv1 = b1b[nb1 + nlane];
#pragma unroll
      for (int mt = 0; mt < 4; ++mt)
#pragma unroll
        for (int r = 0; r < 8; ++r) { acc0[mt][r] = bv0; acc1[mt][r] = bv1; }
    }
    const bf16_t* wrow0 = w1b_t + (size_t)(nb0 + nlane) * 1024;
    const bf16_t* wrow1 = w1b_t + (size_t)(nb1 + nlane) * 1024;
    for (int k = 0; k < 1024; k += 32) {
      const v16bf b0 = load_b_bf16(wrow0 + k, lane);
      const v16bf b1 = load_b_bf16(wrow1 + k, lane);
      v16bf a[4];
#pragma unroll
      for (int mt = 0; mt < 4; ++mt) a[mt] = load_a_bf16(arow_h[mt] + k, lane);
#pragma unroll
      for (int mt = 0; mt < 4; ++mt) acc0[mt] = wmma_bf16(a[mt], b0, acc0[mt]);
#pragma unroll
      for (int mt = 0; mt < 4; ++mt) acc1[mt] = wmma_bf16(a[mt], b1, acc1[mt]);
    }
#pragma unroll
    for (int mt = 0; mt < 4; ++mt)
#pragma unroll
      for (int r = 0; r < 8; ++r) {
        const int mrow = mt * 16 + r + mhi;
        float* dst = sums + (size_t)s_col[mrow] * DD;
        atomicAdd(dst + nb0 + nlane, acc0[mt][r]);
        atomicAdd(dst + nb1 + nlane, acc1[mt][r]);
      }
  }
}

// ---------------------------------------------------------------------------
// Node MLP: out = relu([x | mean] @ W2a + b2a) @ W2b + b2b
// ---------------------------------------------------------------------------
__global__ void __launch_bounds__(256) node_mlp_kernel(
    const bf16_t* __restrict__ xb, const bf16_t* __restrict__ meanb,
    const bf16_t* __restrict__ w2a_t, const float* __restrict__ b2a,
    const bf16_t* __restrict__ w2b_t, const float* __restrict__ b2b,
    float* __restrict__ out) {
  extern __shared__ bf16_t lds[];
  bf16_t* h1   = lds;             // [64][LDH]
  bf16_t* abuf = lds + 64 * LDH;  // [64][ASTR]

  const int tid   = threadIdx.x;
  const int lane  = tid & 31;
  const int wv    = tid >> 5;
  const int nlane = lane & 15;
  const int mhi   = (lane & 16) ? 8 : 0;
  const int n0    = blockIdx.x * 64;

  // ---------------- layer 1 ----------------
  for (int np = 0; np < 4; ++np) {
    const int nb0 = np * 256 + wv * 32;
    const int nb1 = nb0 + 16;
    v8f acc0[4], acc1[4];
    {
      const float bv0 = b2a[nb0 + nlane];
      const float bv1 = b2a[nb1 + nlane];
#pragma unroll
      for (int mt = 0; mt < 4; ++mt)
#pragma unroll
        for (int r = 0; r < 8; ++r) { acc0[mt][r] = bv0; acc1[mt][r] = bv1; }
    }
    const bf16_t* wrow0 = w2a_t + (size_t)(nb0 + nlane) * 2048;
    const bf16_t* wrow1 = w2a_t + (size_t)(nb1 + nlane) * 2048;

    for (int kc = 0; kc < 2048; kc += CK) {
#pragma unroll
      for (int t = 0; t < 4; ++t) {
        const int i   = tid + t * 256;
        const int row = i >> 4;
        const int c   = (i & 15) * 8;
        int grow = n0 + row;
        grow = grow < NN ? grow : NN - 1;  // clamp loads for tail block
        const bf16_t* src = (kc < 1024)
            ? xb + (size_t)grow * DD + kc + c
            : meanb + (size_t)grow * DD + (kc - 1024) + c;
        copy16_g2l(src, abuf + row * ASTR + c);
      }
      async_wait0();
      __syncthreads();

#pragma unroll
      for (int k2 = 0; k2 < CK; k2 += 32) {
        const int k = kc + k2;
        __builtin_prefetch(wrow0 + k + 512, 0, 1);
        __builtin_prefetch(wrow1 + k + 512, 0, 1);
        const v16bf b0 = load_b_bf16(wrow0 + k, lane);
        const v16bf b1 = load_b_bf16(wrow1 + k, lane);
        v16bf a[4];
#pragma unroll
        for (int mt = 0; mt < 4; ++mt)
          a[mt] = load_a_bf16(abuf + (mt * 16 + nlane) * ASTR + k2, lane);
#pragma unroll
        for (int mt = 0; mt < 4; ++mt) acc0[mt] = wmma_bf16(a[mt], b0, acc0[mt]);
#pragma unroll
        for (int mt = 0; mt < 4; ++mt) acc1[mt] = wmma_bf16(a[mt], b1, acc1[mt]);
      }
      __syncthreads();
    }
#pragma unroll
    for (int mt = 0; mt < 4; ++mt)
#pragma unroll
      for (int r = 0; r < 8; ++r) {
        const int mrow = mt * 16 + r + mhi;
        h1[(size_t)mrow * LDH + nb0 + nlane] = (__bf16)fmaxf(acc0[mt][r], 0.0f);
        h1[(size_t)mrow * LDH + nb1 + nlane] = (__bf16)fmaxf(acc1[mt][r], 0.0f);
      }
  }
  __syncthreads();

  // ---------------- layer 2 ----------------
  const bf16_t* arow_h[4];
#pragma unroll
  for (int mt = 0; mt < 4; ++mt)
    arow_h[mt] = h1 + (size_t)(mt * 16 + nlane) * LDH;

  for (int np = 0; np < 4; ++np) {
    const int nb0 = np * 256 + wv * 32;
    const int nb1 = nb0 + 16;
    v8f acc0[4], acc1[4];
    {
      const float bv0 = b2b[nb0 + nlane];
      const float bv1 = b2b[nb1 + nlane];
#pragma unroll
      for (int mt = 0; mt < 4; ++mt)
#pragma unroll
        for (int r = 0; r < 8; ++r) { acc0[mt][r] = bv0; acc1[mt][r] = bv1; }
    }
    const bf16_t* wrow0 = w2b_t + (size_t)(nb0 + nlane) * 1024;
    const bf16_t* wrow1 = w2b_t + (size_t)(nb1 + nlane) * 1024;
    for (int k = 0; k < 1024; k += 32) {
      const v16bf b0 = load_b_bf16(wrow0 + k, lane);
      const v16bf b1 = load_b_bf16(wrow1 + k, lane);
      v16bf a[4];
#pragma unroll
      for (int mt = 0; mt < 4; ++mt) a[mt] = load_a_bf16(arow_h[mt] + k, lane);
#pragma unroll
      for (int mt = 0; mt < 4; ++mt) acc0[mt] = wmma_bf16(a[mt], b0, acc0[mt]);
#pragma unroll
      for (int mt = 0; mt < 4; ++mt) acc1[mt] = wmma_bf16(a[mt], b1, acc1[mt]);
    }
#pragma unroll
    for (int mt = 0; mt < 4; ++mt)
#pragma unroll
      for (int r = 0; r < 8; ++r) {
        const int mrow = mt * 16 + r + mhi;
        const int node = n0 + mrow;
        if (node < NN) {
          float* dst = out + (size_t)node * DD;
          dst[nb0 + nlane] = acc0[mt][r];
          dst[nb1 + nlane] = acc1[mt][r];
        }
      }
  }
}

// ---------------------------------------------------------------------------
// Host launcher
// ---------------------------------------------------------------------------
extern "C" void kernel_launch(void* const* d_in, const int* in_sizes, int n_in,
                              void* d_out, int out_size, void* d_ws,
                              size_t ws_size, hipStream_t stream) {
  const float*     x     = (const float*)d_in[0];
  const long long* ei    = (const long long*)d_in[1];
  const float*     eattr = (const float*)d_in[2];
  const float*     W1a   = (const float*)d_in[3];
  const float*     b1a   = (const float*)d_in[4];
  const float*     W1b   = (const float*)d_in[5];
  const float*     b1b   = (const float*)d_in[6];
  const float*     W2a   = (const float*)d_in[7];
  const float*     b2a   = (const float*)d_in[8];
  const float*     W2b   = (const float*)d_in[9];
  const float*     b2b   = (const float*)d_in[10];
  float* out = (float*)d_out;

  // Workspace layout (bytes)
  constexpr size_t OFF_W1A  = 0;                                   // 4 MB
  constexpr size_t OFF_W1B  = OFF_W1A + (size_t)1024 * 2048 * 2;   // 2 MB
  constexpr size_t OFF_W2A  = OFF_W1B + (size_t)1024 * 1024 * 2;   // 4 MB
  constexpr size_t OFF_W2B  = OFF_W2A + (size_t)1024 * 2048 * 2;   // 2 MB
  constexpr size_t OFF_SUMS = OFF_W2B + (size_t)1024 * 1024 * 2;   // 82 MB f32
  constexpr size_t OFF_CNT  = OFF_SUMS + (size_t)NN * DD * 4;      // 80 KB
  constexpr size_t OFF_XB   = (OFF_CNT + (size_t)NN * 4 + 255) & ~(size_t)255;
  constexpr size_t OFF_EB   = OFF_XB + (size_t)NN * DD * 2;        // 164 MB
  constexpr size_t OFF_MB   = OFF_EB + (size_t)EE * DD * 2;        // 41 MB

  char* ws = (char*)d_ws;
  bf16_t* w1a_t  = (bf16_t*)(ws + OFF_W1A);
  bf16_t* w1b_t  = (bf16_t*)(ws + OFF_W1B);
  bf16_t* w2a_t  = (bf16_t*)(ws + OFF_W2A);
  bf16_t* w2b_t  = (bf16_t*)(ws + OFF_W2B);
  float*  sums   = (float*)(ws + OFF_SUMS);
  float*  counts = (float*)(ws + OFF_CNT);
  bf16_t* xb     = (bf16_t*)(ws + OFF_XB);
  bf16_t* eb     = (bf16_t*)(ws + OFF_EB);
  bf16_t* meanb  = (bf16_t*)(ws + OFF_MB);

  // 64*LDH (h1) + 64*ASTR (staged A) bf16 = 149,504 B -> 2 blocks per WGP
  const int lds_bytes = (64 * LDH + 64 * ASTR) * 2;

  transpose_to_bf16<<<(2048 * 1024 + 255) / 256, 256, 0, stream>>>(W1a, w1a_t, 2048, 1024);
  transpose_to_bf16<<<(1024 * 1024 + 255) / 256, 256, 0, stream>>>(W1b, w1b_t, 1024, 1024);
  transpose_to_bf16<<<(2048 * 1024 + 255) / 256, 256, 0, stream>>>(W2a, w2a_t, 2048, 1024);
  transpose_to_bf16<<<(1024 * 1024 + 255) / 256, 256, 0, stream>>>(W2b, w2b_t, 1024, 1024);

  f32_to_bf16_v4<<<(NN * DD / 4 + 255) / 256, 256, 0, stream>>>(x, xb, NN * DD / 4);
  f32_to_bf16_v4<<<(EE * DD / 4 + 255) / 256, 256, 0, stream>>>(eattr, eb, EE * DD / 4);

  const int zcount = NN * DD + NN;  // sums + counts are contiguous
  zero_f32<<<(zcount + 255) / 256, 256, 0, stream>>>(sums, zcount);

  edge_mlp_kernel<<<EE / 64, 256, lds_bytes, stream>>>(
      xb, ei, eb, w1a_t, b1a, w1b_t, b1b, sums, counts);

  mean_to_bf16<<<(NN * DD + 255) / 256, 256, 0, stream>>>(sums, counts, meanb, NN * DD);

  node_mlp_kernel<<<(NN + 63) / 64, 256, lds_bytes, stream>>>(
      xb, meanb, w2a_t, b2a, w2b_t, b2b, out);
}
